// HybridBlock_57552561766979
// MI455X (gfx1250) — compile-verified, matrix operand order checked by reading, not run on hardware
//
#include <hip/hip_runtime.h>
#include <hip/hip_bf16.h>
#include <stdint.h>

#define DEV __device__ __forceinline__

typedef __bf16 v16bf __attribute__((ext_vector_type(16)));
typedef float  v8f   __attribute__((ext_vector_type(8)));
typedef unsigned int v4u __attribute__((ext_vector_type(4)));
typedef int v8i __attribute__((ext_vector_type(8)));
typedef int v4i __attribute__((ext_vector_type(4)));

#if defined(__gfx1250__) && __has_builtin(__builtin_amdgcn_tensor_load_to_lds) && __has_builtin(__builtin_amdgcn_s_wait_tensorcnt)
#define USE_TDM 1
#else
#define USE_TDM 0
#endif

DEV unsigned short f2bf(float f) {
  unsigned int u = __builtin_bit_cast(unsigned int, f);
  u += 0x7FFFu + ((u >> 16) & 1u);           // round-to-nearest-even
  return (unsigned short)(u >> 16);
}

#if USE_TDM
// Build a 2D Tensor-DMA descriptor (D#) per CDNA5 ISA and
// issue TENSOR_LOAD_TO_LDS (6-arg toolchain form: g0, g1, g2, g3, g4, cpol).
// bf16 elements (data_size=1 -> 2 bytes), no padding, no iteration,
// no multicast (workgroup_mask=0), type=2 ("image").
DEV void tdm_load_tile_2d(unsigned int lds_byte_addr,
                          const unsigned short* gptr_tile,   // tile start in global
                          unsigned int tensor_d0, unsigned int tensor_d1,
                          unsigned int tile_d0, unsigned int tile_d1,
                          unsigned long long row_stride_elems) {
  unsigned long long ga = (unsigned long long)(uintptr_t)gptr_tile;
  v4u g0;
  g0[0] = 1u;                                            // count=1, user mode
  g0[1] = lds_byte_addr;                                 // LDS dest (byte addr)
  g0[2] = (unsigned int)(ga & 0xFFFFFFFFull);            // global_addr[31:0]
  g0[3] = (unsigned int)((ga >> 32) & 0x01FFFFFFull) | (2u << 30); // [56:32] | type=2
  v8i g1;
  g1[0] = (int)(1u << 16);                               // data_size=1 (2B), wg_mask=0
  g1[1] = (int)((tensor_d0 & 0xFFFFu) << 16);            // tensor_dim0[15:0] @ bits63:48
  g1[2] = (int)((tensor_d0 >> 16) | ((tensor_d1 & 0xFFFFu) << 16));
  g1[3] = (int)((tensor_d1 >> 16) | (tile_d0 << 16));    // tile_dim0 @ bits127:112
  g1[4] = (int)(tile_d1 & 0xFFFFu);                      // tile_dim1, tile_dim2=0
  g1[5] = (int)(row_stride_elems & 0xFFFFFFFFull);       // dim0_stride[31:0]
  g1[6] = (int)((row_stride_elems >> 32) & 0xFFFFull);   // dim0_stride[47:32]
  g1[7] = 0;
  v4i z4 = {0, 0, 0, 0};
  v8i z8 = {0, 0, 0, 0, 0, 0, 0, 0};
  __builtin_amdgcn_tensor_load_to_lds(g0, g1, z4, z4, z8, 0);
}
#endif

// ---------------------------------------------------------------------------
// Tiled bf16 WMMA GEMM:  C[M,N] = epilogue( A[M,K] @ W[N,K]^T )
// Block = 256 threads (8 waves), tile 128x128, BK=32 (one wmma K-step).
// Waves arranged 4(M) x 2(N); each wave: 2x4 fragments of 16x16.
// ---------------------------------------------------------------------------
enum { EPI_NONE = 0, EPI_SIGMOID = 1, EPI_RESID = 2, EPI_SQRELU = 3 };

struct Frag32B { uint4 lo, hi; };

template <int EPI>
__global__ __launch_bounds__(256) void gemm_bf16_wmma(
    const unsigned short* __restrict__ A,   // [M,K] bf16
    const unsigned short* __restrict__ W,   // [N,K] bf16 (weight, row = out-ch)
    float* __restrict__ Cf,                 // f32 out (NONE/SIGMOID/RESID)
    unsigned short* __restrict__ Cb,        // bf16 out (SQRELU)
    const float* __restrict__ resid,        // [M,N] residual (RESID)
    const float* __restrict__ scale,        // [N] per-channel scale (RESID)
    int M, int N, int K) {
  constexpr int BM = 128, BN = 128, BK = 32;
  __shared__ __align__(16) unsigned short sA[2][BM * BK];
  __shared__ __align__(16) unsigned short sB[2][BN * BK];

  const int tid = threadIdx.x;
  const int lane = tid & 31;
  const int wid = tid >> 5;
  const int waveM = wid & 3;     // 0..3 -> 32-row strip
  const int waveN = wid >> 2;    // 0..1 -> 64-col strip
  const int r16 = lane & 15;     // row within 16 (A) / col within 16 (B)
  const int hf = lane >> 4;      // K-half select per ISA 16-bit layout

  const int bm = blockIdx.y * BM;
  const int bn = blockIdx.x * BN;
  const int KT = K / BK;

  v8f zero = {0.f, 0.f, 0.f, 0.f, 0.f, 0.f, 0.f, 0.f};
  v8f acc[2][4];
#pragma unroll
  for (int i = 0; i < 2; ++i)
#pragma unroll
    for (int j = 0; j < 4; ++j) acc[i][j] = zero;

#if USE_TDM
  if (tid < 32) {  // single wave drives the DMA (EXEC ignored by TDM)
    tdm_load_tile_2d((unsigned int)(uintptr_t)&sA[0][0], A + (size_t)bm * K,
                     (unsigned int)K, (unsigned int)M, BK, BM, (unsigned long long)K);
    tdm_load_tile_2d((unsigned int)(uintptr_t)&sB[0][0], W + (size_t)bn * K,
                     (unsigned int)K, (unsigned int)N, BK, BN, (unsigned long long)K);
  }
#endif

  for (int kt = 0; kt < KT; ++kt) {
#if USE_TDM
    const int cur = kt & 1;
    __builtin_amdgcn_s_wait_tensorcnt(0);   // wave0: DMA for buf 'cur' done
    __syncthreads();                        // publish buf 'cur'; prev compute done
    if (kt + 1 < KT && tid < 32) {
      const int k0n = (kt + 1) * BK;
      tdm_load_tile_2d((unsigned int)(uintptr_t)&sA[cur ^ 1][0],
                       A + (size_t)bm * K + k0n, (unsigned int)K, (unsigned int)M,
                       BK, BM, (unsigned long long)K);
      tdm_load_tile_2d((unsigned int)(uintptr_t)&sB[cur ^ 1][0],
                       W + (size_t)bn * K + k0n, (unsigned int)K, (unsigned int)N,
                       BK, BN, (unsigned long long)K);
    }
#else
    const int cur = 0;
    const int k0 = kt * BK;
    __syncthreads();
#pragma unroll
    for (int i = 0; i < 2; ++i) {           // cooperative global->LDS staging
      int chunk = tid + i * 256;            // 512 x 16B per tile
      int rr = chunk >> 2;
      int ko = (chunk & 3) * 8;
      *(uint4*)&sA[0][rr * BK + ko] = *(const uint4*)&A[(size_t)(bm + rr) * K + k0 + ko];
      *(uint4*)&sB[0][rr * BK + ko] = *(const uint4*)&W[(size_t)(bn + rr) * K + k0 + ko];
    }
    __syncthreads();
#endif

    // Fragment loads in ISA 16-bit WMMA lane layout:
    // lane<16: K = hf*8+{0..7} (VGPR0-3), K = 16+hf*8+{0..7} (VGPR4-7)
    v16bf af[2], bfm[4];
#pragma unroll
    for (int fm = 0; fm < 2; ++fm) {
      const unsigned short* p = &sA[cur][(waveM * 32 + fm * 16 + r16) * BK + hf * 8];
      Frag32B fr;
      fr.lo = *(const uint4*)p;
      fr.hi = *(const uint4*)(p + 16);
      af[fm] = __builtin_bit_cast(v16bf, fr);
    }
#pragma unroll
    for (int fn = 0; fn < 4; ++fn) {
      const unsigned short* p = &sB[cur][(waveN * 64 + fn * 16 + r16) * BK + hf * 8];
      Frag32B fr;
      fr.lo = *(const uint4*)p;
      fr.hi = *(const uint4*)(p + 16);
      bfm[fn] = __builtin_bit_cast(v16bf, fr);
    }
#pragma unroll
    for (int fm = 0; fm < 2; ++fm)
#pragma unroll
      for (int fn = 0; fn < 4; ++fn)
        acc[fm][fn] = __builtin_amdgcn_wmma_f32_16x16x32_bf16(
            false, af[fm], false, bfm[fn], (short)0, acc[fm][fn], false, false);
  }

  // Epilogue. C layout: lanes0-15 -> M=j, N=lane; lanes16-31 -> M=j+8, N=lane-16.
#pragma unroll
  for (int fm = 0; fm < 2; ++fm) {
#pragma unroll
    for (int fn = 0; fn < 4; ++fn) {
      const int gn = bn + waveN * 64 + fn * 16 + r16;
#pragma unroll
      for (int j = 0; j < 8; ++j) {
        const int gm = bm + waveM * 32 + fm * 16 + j + 8 * hf;
        const float v = acc[fm][fn][j];
        const size_t o = (size_t)gm * N + gn;
        if (EPI == EPI_NONE) {
          Cf[o] = v;
        } else if (EPI == EPI_SIGMOID) {
          Cf[o] = 1.0f / (1.0f + __expf(-v));
        } else if (EPI == EPI_RESID) {
          Cf[o] = resid[o] + scale[gn] * v;
        } else {  // EPI_SQRELU: leaky_relu(0.5) then square -> bf16
          float h = v > 0.0f ? v : 0.5f * v;
          Cb[o] = f2bf(h * h);
        }
      }
    }
  }
}

// ---------------------------------------------------------------------------
// RMS norm over last dim -> bf16.  One block (256 thr = 8 waves) per row.
// ---------------------------------------------------------------------------
__global__ __launch_bounds__(256) void rmsnorm_bf16(const float* __restrict__ x,
                                                    unsigned short* __restrict__ out,
                                                    int Dim) {
  const int row = blockIdx.x;
  const float* xr = x + (size_t)row * Dim;
  float ss = 0.f;
  for (int i = threadIdx.x; i < Dim; i += blockDim.x) {
    float v = xr[i];
    ss += v * v;
  }
  __shared__ float red[8];
  for (int o = 16; o > 0; o >>= 1) ss += __shfl_down(ss, o, 32);
  const int lane = threadIdx.x & 31, wv = threadIdx.x >> 5;
  if (lane == 0) red[wv] = ss;
  __syncthreads();
  if (wv == 0) {
    float v = (lane < 8) ? red[lane] : 0.f;
    for (int o = 4; o > 0; o >>= 1) v += __shfl_down(v, o, 32);
    if (lane == 0) red[0] = v;
  }
  __syncthreads();
  const float inv = rsqrtf(red[0] / (float)Dim + 1e-6f);
  unsigned short* orow = out + (size_t)row * Dim;
  for (int i = threadIdx.x; i < Dim; i += blockDim.x) orow[i] = f2bf(xr[i] * inv);
}

__global__ void cast_f32_bf16(const float* __restrict__ in,
                              unsigned short* __restrict__ out, int n) {
  for (int i = blockIdx.x * blockDim.x + threadIdx.x; i < n; i += gridDim.x * blockDim.x)
    out[i] = f2bf(in[i]);
}

__global__ void mul_inplace(float* __restrict__ a, const float* __restrict__ b, int n) {
  for (int i = blockIdx.x * blockDim.x + threadIdx.x; i < n; i += gridDim.x * blockDim.x)
    a[i] *= b[i];
}

// ---------------------------------------------------------------------------
// Chunked affine scan: s_{t+1} = ew*s_t + kv_t ; wkv_t = s_t + eu*kv_t.
// Pass 1: per-chunk local final states (s0 = 0).
// Pass 2: Horner-combine chunk finals with ew^L to get s_start, emit r*wkv (bf16).
// ---------------------------------------------------------------------------
__global__ void scan_pass1(const float* __restrict__ kv, const float* __restrict__ td,
                           float* __restrict__ finals, int Bb, int S, int Dim, int CH) {
  const int idx = blockIdx.x * blockDim.x + threadIdx.x;  // (c*Bb + b)*Dim + d
  const int d = idx % Dim;
  const int cb = idx / Dim;
  const int b = cb % Bb;
  const int c = cb / Bb;
  const int L = S / CH;
  const float ew = __expf(-__expf(td[d]));
  const float* p = kv + ((size_t)b * S + (size_t)c * L) * Dim + d;
  float s = 0.f;
  for (int t = 0; t < L; ++t) s = ew * s + p[(size_t)t * Dim];
  finals[idx] = s;
}

__global__ void scan_pass2(const float* __restrict__ kv, const float* __restrict__ r,
                           const float* __restrict__ td, const float* __restrict__ tf,
                           const float* __restrict__ finals,
                           unsigned short* __restrict__ rwkv, int Bb, int S, int Dim,
                           int CH) {
  const int idx = blockIdx.x * blockDim.x + threadIdx.x;
  const int d = idx % Dim;
  const int cb = idx / Dim;
  const int b = cb % Bb;
  const int c = cb / Bb;
  const int L = S / CH;
  const float e = __expf(td[d]);
  const float ew = __expf(-e);
  const float ewL = __expf(-(float)L * e);  // ew^L
  const float eu = __expf(tf[d]);
  float s = 0.f;  // state at chunk start
  for (int j = 0; j < c; ++j) s = ewL * s + finals[((size_t)j * Bb + b) * Dim + d];
  const size_t base = ((size_t)b * S + (size_t)c * L) * Dim + d;
  for (int t = 0; t < L; ++t) {
    const size_t o = base + (size_t)t * Dim;
    const float kvt = kv[o];
    const float wkv = s + eu * kvt;         // output uses pre-update state
    rwkv[o] = f2bf(r[o] * wkv);
    s = ew * s + kvt;
  }
}

// ---------------------------------------------------------------------------
extern "C" void kernel_launch(void* const* d_in, const int* in_sizes, int n_in,
                              void* d_out, int out_size, void* d_ws, size_t ws_size,
                              hipStream_t stream) {
  (void)n_in; (void)out_size; (void)ws_size;
  const float* x           = (const float*)d_in[0];
  const float* Wk          = (const float*)d_in[1];
  const float* Wv          = (const float*)d_in[2];
  const float* Wr          = (const float*)d_in[3];
  const float* Wo          = (const float*)d_in[4];
  const float* td          = (const float*)d_in[5];
  const float* tf          = (const float*)d_in[6];
  const float* mixer_scale = (const float*)d_in[7];
  const float* Wfc         = (const float*)d_in[8];
  const float* Wmlp        = (const float*)d_in[9];
  const float* mlp_scale   = (const float*)d_in[10];
  float* out = (float*)d_out;

  const int D = in_sizes[5];            // 1024
  const int H = in_sizes[8] / D;        // 4096
  const int Ntok = in_sizes[0] / D;     // B*S = 16384
  const int Bb = 4;                     // reference batch
  const int S = Ntok / Bb;              // 4096
  const int CH = 16;                    // scan chunks per sequence

  char* ws = (char*)d_ws;
  size_t off = 0;
  auto alloc = [&](size_t bytes) {
    char* p = ws + off;
    off += (bytes + 255) & ~(size_t)255;
    return p;
  };
  unsigned short* wk_b   = (unsigned short*)alloc((size_t)D * D * 2);
  unsigned short* wv_b   = (unsigned short*)alloc((size_t)D * D * 2);
  unsigned short* wr_b   = (unsigned short*)alloc((size_t)D * D * 2);
  unsigned short* wo_b   = (unsigned short*)alloc((size_t)D * D * 2);
  unsigned short* wfc_b  = (unsigned short*)alloc((size_t)H * D * 2);
  unsigned short* wmlp_b = (unsigned short*)alloc((size_t)D * H * 2);
  unsigned short* n1     = (unsigned short*)alloc((size_t)Ntok * D * 2);  // normed (reused for 2nd norm)
  float* bk     = (float*)alloc((size_t)Ntok * D * 4);  // k, then k*v
  float* bv     = (float*)alloc((size_t)Ntok * D * 4);  // v   (contiguous after bk)
  float* br     = (float*)alloc((size_t)Ntok * D * 4);  // sigmoid(r)
  float* finals = (float*)alloc((size_t)CH * Bb * D * 4);
  unsigned short* rwkv = (unsigned short*)alloc((size_t)Ntok * D * 2);
  float* x1     = (float*)alloc((size_t)Ntok * D * 4);
  // hh [Ntok,H] bf16 aliases bk+bv (kv dead after scan; 64MB+64MB = 128MB)
  unsigned short* hh = (unsigned short*)bk;

  // 1) weights -> bf16 (weights all fit in L2; converted once per call)
  cast_f32_bf16<<<1024, 256, 0, stream>>>(Wk, wk_b, D * D);
  cast_f32_bf16<<<1024, 256, 0, stream>>>(Wv, wv_b, D * D);
  cast_f32_bf16<<<1024, 256, 0, stream>>>(Wr, wr_b, D * D);
  cast_f32_bf16<<<1024, 256, 0, stream>>>(Wo, wo_b, D * D);
  cast_f32_bf16<<<2048, 256, 0, stream>>>(Wfc, wfc_b, H * D);
  cast_f32_bf16<<<2048, 256, 0, stream>>>(Wmlp, wmlp_b, D * H);

  // 2) RMS-norm(x) -> bf16
  rmsnorm_bf16<<<Ntok, 256, 0, stream>>>(x, n1, D);

  // 3) k, v, sigmoid(r) GEMMs
  dim3 gD(D / 128, Ntok / 128);
  gemm_bf16_wmma<EPI_NONE><<<gD, 256, 0, stream>>>(n1, wk_b, bk, nullptr, nullptr, nullptr, Ntok, D, D);
  gemm_bf16_wmma<EPI_NONE><<<gD, 256, 0, stream>>>(n1, wv_b, bv, nullptr, nullptr, nullptr, Ntok, D, D);
  gemm_bf16_wmma<EPI_SIGMOID><<<gD, 256, 0, stream>>>(n1, wr_b, br, nullptr, nullptr, nullptr, Ntok, D, D);

  // 4) kv = k*v (in-place into bk)
  mul_inplace<<<2048, 256, 0, stream>>>(bk, bv, Ntok * D);

  // 5) chunked scan -> rwkv = sigmoid(r)*wkv in bf16
  const int scan_threads = CH * Bb * D;
  scan_pass1<<<scan_threads / 256, 256, 0, stream>>>(bk, td, finals, Bb, S, D, CH);
  scan_pass2<<<scan_threads / 256, 256, 0, stream>>>(bk, br, td, tf, finals, rwkv, Bb, S, D, CH);

  // 6) mixer out projection + residual: x1 = x + mixer_scale * (rwkv @ Wo^T)
  gemm_bf16_wmma<EPI_RESID><<<gD, 256, 0, stream>>>(rwkv, wo_b, x1, nullptr, x, mixer_scale, Ntok, D, D);

  // 7) MLP: hh = leaky_relu(RMS(x1) @ Wfc^T, 0.5)^2  (bf16)
  rmsnorm_bf16<<<Ntok, 256, 0, stream>>>(x1, n1, D);
  dim3 gH(H / 128, Ntok / 128);
  gemm_bf16_wmma<EPI_SQRELU><<<gH, 256, 0, stream>>>(n1, wfc_b, nullptr, hh, nullptr, nullptr, Ntok, H, D);

  // 8) out = x1 + mlp_scale * (hh @ Wmlp^T)
  gemm_bf16_wmma<EPI_RESID><<<gD, 256, 0, stream>>>(hh, wmlp_b, out, nullptr, x1, mlp_scale, Ntok, D, H);
}
